// SobaMonarch_7481833030020
// MI455X (gfx1250) — matrix operand order repaired and checked by reading, not compiled, not run on hardware
//
#include <hip/hip_runtime.h>

typedef __attribute__((ext_vector_type(16))) _Float16 v16h;
typedef __attribute__((ext_vector_type(8)))  _Float16 v8h;
typedef __attribute__((ext_vector_type(4)))  _Float16 v4h;
typedef __attribute__((ext_vector_type(8)))  float    v8f;
typedef __attribute__((ext_vector_type(4)))  float    v4f;

#define LDH 72        // f16 LDS row stride (halfs): 144B row -> 16B-aligned frags, conflict-free columns
#define LDF 65        // f32 LDS row stride
#define HB  262144    // 64*64*64 elements per head (intermediates)
#define SD  262144    // 4096*64 elements per head (Q/K/V/out)

// ---------------- WMMA fragment helpers (gfx1250 wave32 layouts) ----------------
// A: 16x32 f16 row-major (M,K). lanes 0-15: M=lane, K: v0..3->0..7, v4..7->16..23
//                               lanes 16-31: M=lane-16, K: 8..15 and 24..31
__device__ inline v16h frag_a(const _Float16* M, int ld, int m0, int k0) {
  int lane = threadIdx.x & 31;
  int m  = m0 + (lane & 15);
  int kb = k0 + ((lane >> 4) << 3);
  const _Float16* p = M + m * ld + kb;      // (m*144 + kb*2) % 16 == 0 -> b128 loads
  v8h lo = *(const v8h*)p;
  v8h hi = *(const v8h*)(p + 16);
  v16h f;
#pragma unroll
  for (int v = 0; v < 8; ++v) { f[v] = lo[v]; f[8 + v] = hi[v]; }
  return f;
}
// B operand given B^T stored row-major (N,K): element (k,n) = Bt[n*ld + k], same K order as frag_a
__device__ inline v16h frag_bT(const _Float16* Bt, int ld, int k0, int n0) {
  int lane = threadIdx.x & 31;
  int n  = n0 + (lane & 15);
  int kb = k0 + ((lane >> 4) << 3);
  const _Float16* p = Bt + n * ld + kb;     // 16B aligned -> 2x b128
  v8h lo = *(const v8h*)p;
  v8h hi = *(const v8h*)(p + 16);
  v16h f;
#pragma unroll
  for (int v = 0; v < 8; ++v) { f[v] = lo[v]; f[8 + v] = hi[v]; }
  return f;
}
// C/D f32 16x16: VGPR r: lanes 0-15 -> (M=r, N=lane); lanes 16-31 -> (M=8+r, N=lane-16)
__device__ inline void store_c_f32(float* C, int ld, int m0, int n0, v8f acc) {
  int lane = threadIdx.x & 31;
  int n  = n0 + (lane & 15);
  int mb = m0 + ((lane >> 4) << 3);
#pragma unroll
  for (int r = 0; r < 8; ++r) C[(size_t)(mb + r) * ld + n] = acc[r];
}
__device__ inline void store_c_f16(_Float16* C, int ld, int m0, int n0, v8f acc) {
  int lane = threadIdx.x & 31;
  int n  = n0 + (lane & 15);
  int mb = m0 + ((lane >> 4) << 3);
#pragma unroll
  for (int r = 0; r < 8; ++r) C[(size_t)(mb + r) * ld + n] = (_Float16)acc[r];
}

__device__ inline v8f wmma32(v16h a, v16h b, v8f c) {
  return __builtin_amdgcn_wmma_f32_16x16x32_f16(false, a, false, b, (short)0, c, false, false);
}

// 64x64x64 matmul: C = A(64xK) @ B(Kx64), A row-major, B supplied transposed (N,K).
// 8 waves x 2 tiles sharing tm; 4 WMMAs interleaved across 2 accumulators.
template<bool OUT16, typename CT>
__device__ inline void mm64(const _Float16* A, int lda, const _Float16* Bt, int ldb,
                            CT* Cp, int ldc) {
  int wave = threadIdx.x >> 5;
  int tm  = (wave >> 1) << 4;
  int tn0 = (wave & 1) << 5;       // 0 or 32
  int tn1 = tn0 + 16;
  v16h a0  = frag_a (A,  lda, tm, 0);
  v16h a1  = frag_a (A,  lda, tm, 32);
  v16h b00 = frag_bT(Bt, ldb, 0,  tn0);
  v16h b10 = frag_bT(Bt, ldb, 0,  tn1);
  v16h b01 = frag_bT(Bt, ldb, 32, tn0);
  v16h b11 = frag_bT(Bt, ldb, 32, tn1);
  v8f c0 = {}, c1 = {};
  c0 = wmma32(a0, b00, c0);
  c1 = wmma32(a0, b10, c1);
  c0 = wmma32(a1, b01, c0);
  c1 = wmma32(a1, b11, c1);
  if (OUT16) { store_c_f16((_Float16*)Cp, ldc, tm, tn0, c0); store_c_f16((_Float16*)Cp, ldc, tm, tn1, c1); }
  else       { store_c_f32((float*)Cp,    ldc, tm, tn0, c0); store_c_f32((float*)Cp,    ldc, tm, tn1, c1); }
}

// ---------------- staging: 64x64 global tile -> LDS (f16), row-major or transposed ----------------
__device__ inline void stage_f32_rm(const float* g, int grs, _Float16* d, float scale) {
  for (int idx = threadIdx.x; idx < 1024; idx += 256) {
    int r = idx >> 4, c = (idx & 15) << 2;
    v4f x = *(const v4f*)(g + (size_t)r * grs + c);
    v4h h;
#pragma unroll
    for (int u = 0; u < 4; ++u) h[u] = (_Float16)(x[u] * scale);
    *(v4h*)(d + r * LDH + c) = h;
  }
}
__device__ inline void stage_f32_tr(const float* g, int grs, _Float16* d, float scale) {
  for (int idx = threadIdx.x; idx < 1024; idx += 256) {
    int r = idx >> 4, c = (idx & 15) << 2;
    v4f x = *(const v4f*)(g + (size_t)r * grs + c);
#pragma unroll
    for (int u = 0; u < 4; ++u) d[(c + u) * LDH + r] = (_Float16)(x[u] * scale);
  }
}
__device__ inline void stage_f32_rm_tr(const float* g, int grs, _Float16* drm, _Float16* dtr) {
  for (int idx = threadIdx.x; idx < 1024; idx += 256) {
    int r = idx >> 4, c = (idx & 15) << 2;
    v4f x = *(const v4f*)(g + (size_t)r * grs + c);
    v4h h;
#pragma unroll
    for (int u = 0; u < 4; ++u) h[u] = (_Float16)x[u];
    *(v4h*)(drm + r * LDH + c) = h;
#pragma unroll
    for (int u = 0; u < 4; ++u) dtr[(c + u) * LDH + r] = h[u];
  }
}
__device__ inline void stage_f16_rm(const _Float16* g, int grs, _Float16* d) {
  for (int idx = threadIdx.x; idx < 1024; idx += 256) {
    int r = idx >> 4, c = (idx & 15) << 2;
    *(v4h*)(d + r * LDH + c) = *(const v4h*)(g + (size_t)r * grs + c);
  }
}
__device__ inline void stage_f16_tr(const _Float16* g, int grs, _Float16* d) {
  for (int idx = threadIdx.x; idx < 1024; idx += 256) {
    int r = idx >> 4, c = (idx & 15) << 2;
    v4h x = *(const v4h*)(g + (size_t)r * grs + c);
#pragma unroll
    for (int u = 0; u < 4; ++u) d[(c + u) * LDH + r] = x[u];
  }
}

// ---------------- kernels ----------------
__global__ __launch_bounds__(256) void soba_init_left(_Float16* left) {
  v4h v; v[0] = v[1] = v[2] = v[3] = (_Float16)0.015625f;   // 1/64
  size_t base = (size_t)blockIdx.x * 4096 + (size_t)threadIdx.x * 4;
#pragma unroll
  for (int r = 0; r < 4; ++r) *(v4h*)(left + base + (size_t)r * 1024) = v;
}

// per (head, j): Qhat[j](k,v) = left[j]^T @ (Q[:,j,:]/8); tau[j][k] = colsum(left[j])
__global__ __launch_bounds__(256) void soba_qhat(const float* __restrict__ Q,
                                                 const _Float16* __restrict__ left,
                                                 _Float16* __restrict__ qhat,
                                                 float* __restrict__ tau) {
  int j = blockIdx.x, head = blockIdx.y;
  __shared__ __align__(16) _Float16 sLT[64 * LDH];  // (k,l) = left[j]^T
  __shared__ __align__(16) _Float16 sQT[64 * LDH];  // (v,l) = Qj^T  (B^T for B=(l,v))
  const _Float16* lb = left + (size_t)head * HB + (size_t)j * 4096;  // rows l, stride 64
  const float*    qb = Q    + (size_t)head * SD + (size_t)j * 64;    // rows l, stride 4096
  stage_f16_tr(lb, 64, sLT);
  stage_f32_tr(qb, 4096, sQT, 0.125f);
  __syncthreads();
  if (threadIdx.x < 64) {
    float s = 0.f;
    for (int l = 0; l < 64; ++l) s += (float)sLT[threadIdx.x * LDH + l];
    tau[(size_t)head * 4096 + j * 64 + threadIdx.x] = s;
  }
  mm64<true>(sLT, LDH, sQT, LDH, qhat + (size_t)head * HB + (size_t)j * 64, 4096);
}

// per (head, k): beta = Qhat(:,k,:) @ K[k]^T ; softmax(beta/tau) -> right, alpha ;
//                Khat[k] = right @ K[k]
__global__ __launch_bounds__(256) void soba_right(const float* __restrict__ K,
                                                  const _Float16* __restrict__ qhat,
                                                  const float* __restrict__ tau,
                                                  _Float16* __restrict__ right,
                                                  float* __restrict__ alpha,
                                                  _Float16* __restrict__ khat) {
  int k = blockIdx.x, head = blockIdx.y;
  __shared__ __align__(16) _Float16 sK [64 * LDH];  // (i,v)  = B^T for beta
  __shared__ __align__(16) _Float16 sKT[64 * LDH];  // (v,i)  = B^T for Khat
  __shared__ __align__(16) _Float16 sQh[64 * LDH];  // (j,v)
  __shared__ __align__(16) _Float16 sR [64 * LDH];  // (j,i)
  __shared__ __align__(16) float    sB [64 * LDF];  // beta (j,i)
  const float*    kb = K    + (size_t)head * SD + (size_t)k * 4096;
  const _Float16* qh = qhat + (size_t)head * HB + (size_t)k * 4096;
  stage_f32_rm_tr(kb, 64, sK, sKT);
  stage_f16_rm(qh, 64, sQh);
  __syncthreads();
  mm64<false>(sQh, LDH, sK, LDH, sB, LDF);          // beta(j,i) = Qh(j,v) . K(i,v)
  __syncthreads();
  if (threadIdx.x < 64) {
    int j = threadIdx.x;
    float tv  = tau[(size_t)head * 4096 + j * 64 + k];
    float inv = (fabsf(tv) < 1e-30f) ? 1.0f : 1.0f / tv;
    float* row = sB + j * LDF;
    float m = -3.402823466e38f;
    for (int i = 0; i < 64; ++i) { float x = row[i] * inv; row[i] = x; m = fmaxf(m, x); }
    float s = 0.f;
    for (int i = 0; i < 64; ++i) { float e = __expf(row[i] - m); row[i] = e; s += e; }
    float rs = 1.0f / s, al = 0.f;
    _Float16* lr = sR + j * LDH;
    _Float16* gr = right + (size_t)head * HB + (size_t)k * 4096 + j * 64;
    for (int i = 0; i < 64; ++i) {
      float r = row[i] * rs;
      al += (r > 0.f) ? r * __logf(r) : 0.f;
      _Float16 h = (_Float16)r;
      lr[i] = h; gr[i] = h;
    }
    alpha[(size_t)head * 4096 + k * 64 + j] = al;
  }
  __syncthreads();
  mm64<true>(sR, LDH, sKT, LDH, khat + (size_t)head * HB + (size_t)k * 4096, 64);
}

// per (head, j): beta2 = (Q[:,j,:]/8) @ Khat(:,j,:)^T ; left[j] = softmax(beta2 - alpha)
__global__ __launch_bounds__(256) void soba_left(const float* __restrict__ Q,
                                                 const _Float16* __restrict__ khat,
                                                 const float* __restrict__ alpha,
                                                 _Float16* __restrict__ left) {
  int j = blockIdx.x, head = blockIdx.y;
  __shared__ __align__(16) _Float16 sQ [64 * LDH];  // (l,v)
  __shared__ __align__(16) _Float16 sKh[64 * LDH];  // (k,v) = B^T for beta2
  __shared__ __align__(16) float    sB [64 * LDF];  // beta2 (l,k)
  __shared__ float sAl[64];
  const float*    qb = Q    + (size_t)head * SD + (size_t)j * 64;
  const _Float16* kh = khat + (size_t)head * HB + (size_t)j * 64;   // rows k, stride 4096
  stage_f32_rm(qb, 4096, sQ, 0.125f);
  stage_f16_rm(kh, 4096, sKh);
  if (threadIdx.x < 64)
    sAl[threadIdx.x] = alpha[(size_t)head * 4096 + threadIdx.x * 64 + j];
  __syncthreads();
  mm64<false>(sQ, LDH, sKh, LDH, sB, LDF);          // beta2(l,k) = Q(l,v) . Khat(k,v)
  __syncthreads();
  if (threadIdx.x < 64) {
    int l = threadIdx.x;
    float* row = sB + l * LDF;
    float m = -3.402823466e38f;
    for (int kk = 0; kk < 64; ++kk) { float x = row[kk] - sAl[kk]; row[kk] = x; m = fmaxf(m, x); }
    float s = 0.f;
    for (int kk = 0; kk < 64; ++kk) { float e = __expf(row[kk] - m); row[kk] = e; s += e; }
    float rs = 1.0f / s;
    _Float16* gl = left + (size_t)head * HB + (size_t)j * 4096 + l * 64;
    for (int kk = 0; kk < 64; ++kk) gl[kk] = (_Float16)(row[kk] * rs);
  }
}

// per (head, k): Y[k](j,v) = right[k] @ V[k]
__global__ __launch_bounds__(256) void soba_y(const float* __restrict__ V,
                                              const _Float16* __restrict__ right,
                                              _Float16* __restrict__ Y) {
  int k = blockIdx.x, head = blockIdx.y;
  __shared__ __align__(16) _Float16 sVT[64 * LDH];  // (v,i) = V^T
  __shared__ __align__(16) _Float16 sR [64 * LDH];  // (j,i)
  const float*    vb = V     + (size_t)head * SD + (size_t)k * 4096;
  const _Float16* rb = right + (size_t)head * HB + (size_t)k * 4096;
  stage_f32_tr(vb, 64, sVT, 1.0f);
  stage_f16_rm(rb, 64, sR);
  __syncthreads();
  mm64<true>(sR, LDH, sVT, LDH, Y + (size_t)head * HB + (size_t)k * 4096, 64);
}

// per (head, j): Z(l,v) = left[j] @ Y(:,j,:) -> out rows s = l*64+j
__global__ __launch_bounds__(256) void soba_z(const _Float16* __restrict__ left,
                                              const _Float16* __restrict__ Y,
                                              float* __restrict__ out) {
  int j = blockIdx.x, head = blockIdx.y;
  __shared__ __align__(16) _Float16 sL [64 * LDH];  // (l,k)
  __shared__ __align__(16) _Float16 sYT[64 * LDH];  // (v,k) = Y^T
  const _Float16* lb = left + (size_t)head * HB + (size_t)j * 4096;
  const _Float16* yb = Y    + (size_t)head * HB + (size_t)j * 64;   // rows k, stride 4096
  stage_f16_rm(lb, 64, sL);
  stage_f16_tr(yb, 4096, sYT);
  __syncthreads();
  mm64<false>(sL, LDH, sYT, LDH, out + (size_t)head * SD + (size_t)j * 64, 4096);
}

extern "C" void kernel_launch(void* const* d_in, const int* in_sizes, int n_in,
                              void* d_out, int out_size, void* d_ws, size_t ws_size,
                              hipStream_t stream) {
  (void)in_sizes; (void)n_in; (void)out_size; (void)ws_size;
  const float* Q = (const float*)d_in[0];
  const float* K = (const float*)d_in[1];
  const float* V = (const float*)d_in[2];
  float* out = (float*)d_out;

  // f16 workspace: left | right | qhat (reused as Y) | khat ; then f32 tau | alpha
  _Float16* left  = (_Float16*)d_ws;
  _Float16* right = left  + (size_t)64 * HB;
  _Float16* qhat  = right + (size_t)64 * HB;
  _Float16* khat  = qhat  + (size_t)64 * HB;
  float* tau   = (float*)(khat + (size_t)64 * HB);
  float* alpha = tau + (size_t)64 * 4096;

  dim3 grid(64, 64), block(256);
  soba_init_left<<<4096, 256, 0, stream>>>(left);
  for (int step = 0; step < 3; ++step) {
    soba_qhat <<<grid, block, 0, stream>>>(Q, left, qhat, tau);
    soba_right<<<grid, block, 0, stream>>>(K, qhat, tau, right, alpha, khat);
    soba_left <<<grid, block, 0, stream>>>(Q, khat, alpha, left);
  }
  soba_y<<<grid, block, 0, stream>>>(V, right, qhat /* Y reuses qhat */);
  soba_z<<<grid, block, 0, stream>>>(left, qhat, out);
}
